// CliffordEquivariantUpdate_86603720556976
// MI455X (gfx1250) — matrix operand order repaired and checked by reading, not compiled, not run on hardware
//
#include <hip/hip_runtime.h>
#include <math.h>

// ---------------------------------------------------------------------------
// Clifford equivariant GNN update for MI455X (gfx1250, wave32, WMMA).
// One wave = one 16-edge (or 16-node) tile. All channel-mixing linears run on
// v_wmma_f32_16x16x32_f16 (M=16 edges, N=8 out-ch padded to 16, K=16/8 in-ch
// padded to 32). All blade-local math (SiLU, grade-norms, Cayley geometric
// product, layernorm) is lane-local in the WMMA C layout (lane = channel,
// VGPR row = edge).
//
// Data movement: h (6.4 MB) is L2-resident (192 MB L2) -> the kernel is
// L2-bandwidth/atomic bound. Layer weights (~1.4K floats) are staged once per
// block into LDS with bank-conflict-free padded strides; per-wave B-fragment
// builds then hit LDS instead of re-reading global memory. Aggregation is
// f32 global atomics into an L2-resident accumulator in the workspace.
// ---------------------------------------------------------------------------

typedef __attribute__((ext_vector_type(16))) _Float16 v16h;
typedef __attribute__((ext_vector_type(8)))  float    v8f;

#define FCH  8
#define NBL  8
#define FCHP 9          // padded channel stride for the LDS transpose buffer
#define EPSV 1e-6f

// ----- LDS weight-buffer layout (padded strides -> distinct banks/lane) -----
#define OFF_LIN  0      // [8][16][4] stride 68
#define OFF_LINB 544    // [8]
#define OFF_SA   552    // [8][4]
#define OFF_SB   584    // [8][4]
#define OFF_WR   616    // [8][8][4] stride 36
#define OFF_PW   904    // [8][20]   stride 24
#define OFF_WL   1096   // [8][8][4] stride 36
#define OFF_GBL  1384   // [8]
#define OFF_AN   1392   // [8][4]
#define OFF_LNA  1424   // [8]
#define WBUF_SZ  1432

// ----- compile-time Cayley / path tables (mirrors the reference builder) ----
struct GeoTabs {
  int   j[8][8];        // output blade index of blade_i * blade_k
  float s[8][8];        // sign of blade_i * blade_k (euclidean metric)
  int   p[4][4][4];     // (grade_i, grade_j, grade_k) -> path weight index
};

constexpr int kBM[8] = {0, 1, 2, 4, 3, 5, 6, 7};   // shortlex blade bitmaps

constexpr int popc3(int x) { return (x & 1) + ((x >> 1) & 1) + ((x >> 2) & 1); }

constexpr int signpar(int a, int b) {
  int s = 0; int aa = a >> 1;
  while (aa) { s += popc3(aa & b); aa >>= 1; }
  return s & 1;
}

constexpr GeoTabs make_tabs() {
  GeoTabs t{};
  bool used[4][4][4] = {};
  for (int i = 0; i < 8; ++i)
    for (int k = 0; k < 8; ++k) {
      int bi = kBM[i], bk = kBM[k];
      int bj = bi ^ bk;
      int j  = kBM[bj];                 // bitmap->index (involution)
      t.j[i][k] = j;
      t.s[i][k] = signpar(bi, bk) ? -1.0f : 1.0f;
      used[popc3(bi)][popc3(bj)][popc3(bk)] = true;
    }
  int c = 0;
  for (int a = 0; a < 4; ++a)
    for (int b = 0; b < 4; ++b)
      for (int d = 0; d < 4; ++d)
        t.p[a][b][d] = used[a][b][d] ? (c++) : 0;
  return t;
}

// ----- WMMA helper ----------------------------------------------------------
__device__ inline v8f wmma16(v16h a, v16h b, v8f c) {
  // (neg_a, A, neg_b, B, c_mod, C, reuse_a, reuse_b)
  return __builtin_amdgcn_wmma_f32_16x16x32_f16(false, a, false, b,
                                                (short)0, c, false, false);
}

// B fragment from the LDS weight buffer: B[k=m, n=lane] = w[n, m, grade];
// elements m < INDIM are filled from base, the rest (and all of an inactive lane) are 0.
template <int INDIM, int NSTRIDE>
__device__ inline v16h load_bfrag_lds(const float* wb, int base, int n, int g,
                                      bool active) {
  v16h b;
#pragma unroll
  for (int m = 0; m < 16; ++m) b[m] = (_Float16)0.0f;
  if (active) {
#pragma unroll
    for (int m = 0; m < INDIM; ++m)
      b[m] = (_Float16)wb[base + n * NSTRIDE + m * 4 + g];
  }
  return b;
}

__global__ void zero_f32_kernel(float* __restrict__ p, long n) {
  long i = (long)blockIdx.x * blockDim.x + threadIdx.x;
  long stride = (long)gridDim.x * blockDim.x;
  for (; i < n; i += stride) p[i] = 0.0f;
}

// ----- fused CEMLP block: MODE 0 = edge model (atomic agg), 1 = node model --
template <int MODE>
__global__ void __launch_bounds__(256) clifford_block_kernel(
    const float* __restrict__ hsrc,
    const int*   __restrict__ row,
    const int*   __restrict__ col,
    const float* __restrict__ src2,     // node mode: agg features
    const float* __restrict__ lin_w,  const float* __restrict__ lin_b,
    const float* __restrict__ silu_a, const float* __restrict__ silu_b,
    const float* __restrict__ gp_wr,  const float* __restrict__ gp_w,
    const float* __restrict__ gp_wl,  const float* __restrict__ gp_bl,
    const float* __restrict__ gp_an,  const float* __restrict__ ln_a,
    float* __restrict__ outp, int n_items)
{
  constexpr GeoTabs TT = make_tabs();
  constexpr int GR[8] = {0, 1, 1, 1, 2, 2, 2, 3};

  __shared__ float ylds[8][NBL][16][FCHP];  // per-wave post-SiLU staging
  __shared__ float wbuf[WBUF_SZ];           // block-shared layer weights

  const int tid = threadIdx.x;

  // ---- cooperative weight staging (once per block, coalesced) -------------
  for (int i = tid; i < 512; i += 256) wbuf[OFF_LIN + (i >> 6) * 68 + (i & 63)] = lin_w[i];
  for (int i = tid; i < 256; i += 256) wbuf[OFF_WR  + (i >> 5) * 36 + (i & 31)] = gp_wr[i];
  for (int i = tid; i < 256; i += 256) wbuf[OFF_WL  + (i >> 5) * 36 + (i & 31)] = gp_wl[i];
  for (int i = tid; i < 160; i += 256) wbuf[OFF_PW  + (i / 20) * 24 + (i % 20)] = gp_w[i];
  if (tid < 32) {
    wbuf[OFF_SA + tid] = silu_a[tid];
    wbuf[OFF_SB + tid] = silu_b[tid];
    wbuf[OFF_AN + tid] = gp_an[tid];
  } else if (tid < 64) {
    const int k = tid - 32;
    if (k < 8)       wbuf[OFF_LINB + k]      = lin_b[k];
    else if (k < 16) wbuf[OFF_GBL + (k - 8)] = gp_bl[k - 8];
    else if (k < 24) wbuf[OFF_LNA + (k - 16)] = ln_a[k - 16];
  }
  __syncthreads();

  const int lane = tid & 31;
  const int wave = tid >> 5;
  const int tile = blockIdx.x * 8 + wave;
  const int half = lane >> 4;
  const int sub  = lane & 15;
  const int nch  = sub;                 // this lane's output channel (N)
  const int ns   = nch & 7;             // safe param index
  const bool chan_ok = (nch < FCH);
  const bool bact = (half == 0) && chan_ok;
  const long e0 = (long)tile * 16;

  __builtin_prefetch(hsrc, 0, 0);       // global_prefetch_b8

  // ---- gather: lanes 0-15 supply input channels 0-7, lanes 16-31 ch 8-15 --
  long item = e0 + sub;
  if (item > (long)n_items - 1) item = (long)n_items - 1;   // clamp tail
  const float* fp;
  if (MODE == 0) {
    const int node = (half == 0) ? row[item] : col[item];
    fp = hsrc + (long)node * (FCH * NBL);
  } else {
    fp = (half == 0) ? (hsrc + item * (FCH * NBL))
                     : (src2 + item * (FCH * NBL));
  }
  float hv[64] __attribute__((aligned(16)));
#pragma unroll
  for (int q = 0; q < 16; ++q)
    ((float4*)hv)[q] = ((const float4*)fp)[q];

  // A fragments (per blade): elements 0..7 = this lane's 8 input channels
  v16h a1[NBL];
#pragma unroll
  for (int bl = 0; bl < NBL; ++bl) {
#pragma unroll
    for (int m = 0; m < 16; ++m) a1[bl][m] = (_Float16)0.0f;
#pragma unroll
    for (int ch = 0; ch < 8; ++ch) a1[bl][ch] = (_Float16)hv[ch * NBL + bl];
  }

  v8f zacc;
#pragma unroll
  for (int r = 0; r < 8; ++r) zacc[r] = 0.0f;

  // ---- mv_linear 2F -> F (8 WMMAs, one per blade, grade-tied B) -----------
  v8f cl[NBL];
#pragma unroll
  for (int bl = 0; bl < NBL; ++bl) {
    v16h b = load_bfrag_lds<8, 68>(wbuf, OFF_LIN + half * 32, ns, GR[bl], chan_ok);
    cl[bl] = wmma16(a1[bl], b, zacc);
  }

  // ---- MVSiLU (lane-local: lane holds all 8 blades for 8 edges) -----------
  float sa[4], sb[4];
#pragma unroll
  for (int g = 0; g < 4; ++g) {
    sa[g] = wbuf[OFF_SA + ns * 4 + g];
    sb[g] = wbuf[OFF_SB + ns * 4 + g];
  }
  const float linb = wbuf[OFF_LINB + ns];

  float y[NBL][8];
#pragma unroll
  for (int r = 0; r < 8; ++r) {
    float x[NBL];
#pragma unroll
    for (int bl = 0; bl < NBL; ++bl) x[bl] = cl[bl][r];
    x[0] += linb;                                       // bias on scalar blade
    const float inv0 = x[0];
    const float inv1 = x[1]*x[1] + x[2]*x[2] + x[3]*x[3];
    const float inv2 = x[4]*x[4] + x[5]*x[5] + x[6]*x[6];
    const float inv3 = x[7]*x[7];
    const float invq[4] = {inv0, inv1, inv2, inv3};
    float gate[4];
#pragma unroll
    for (int g = 0; g < 4; ++g)
      gate[g] = 1.0f / (1.0f + __expf(-(sa[g]*invq[g] + sb[g])));
#pragma unroll
    for (int bl = 0; bl < NBL; ++bl) y[bl][r] = gate[GR[bl]] * x[bl];
  }

  // ---- transpose post-SiLU through LDS for the next WMMA A fragments ------
  if (chan_ok) {
#pragma unroll
    for (int bl = 0; bl < NBL; ++bl)
#pragma unroll
      for (int r = 0; r < 8; ++r)
        ylds[wave][bl][half * 8 + r][nch] = y[bl][r];
  }
  asm volatile("s_wait_dscnt 0" ::: "memory");  // intra-wave LDS visibility

  v16h a2[NBL];
#pragma unroll
  for (int bl = 0; bl < NBL; ++bl) {
#pragma unroll
    for (int m = 0; m < 16; ++m) a2[bl][m] = (_Float16)0.0f;
    if (half == 0) {                    // K=0..7 live, K=8..31 zero padding
#pragma unroll
      for (int m = 0; m < 8; ++m)
        a2[bl][m] = (_Float16)ylds[wave][bl][sub][m];
    }
  }

  // ---- sgp right/left linears (16 WMMAs) ----------------------------------
  v8f cr[NBL], cw[NBL];
#pragma unroll
  for (int bl = 0; bl < NBL; ++bl) {
    v16h bwr = load_bfrag_lds<8, 36>(wbuf, OFF_WR, ns, GR[bl], bact);
    cr[bl] = wmma16(a2[bl], bwr, zacc);
    v16h bwl = load_bfrag_lds<8, 36>(wbuf, OFF_WL, ns, GR[bl], bact);
    cw[bl] = wmma16(a2[bl], bwl, zacc);
  }

  // ---- normalization + geometric product + layernorm + store --------------
  float anp[4];
#pragma unroll
  for (int g = 0; g < 4; ++g)
    anp[g] = 1.0f / (1.0f + __expf(-wbuf[OFF_AN + ns * 4 + g]));
  float pw[20];
#pragma unroll
  for (int p = 0; p < 20; ++p) pw[p] = wbuf[OFF_PW + ns * 24 + p];
  const float gblb  = wbuf[OFF_GBL + ns];
  const float lna   = wbuf[OFF_LNA + ns];
  const float invs2 = 0.70710678118654752f;

#pragma unroll
  for (int r = 0; r < 8; ++r) {
    float xr[NBL];
#pragma unroll
    for (int bl = 0; bl < NBL; ++bl) xr[bl] = cr[bl][r];
    const float q0 = xr[0]*xr[0];
    const float q1 = xr[1]*xr[1] + xr[2]*xr[2] + xr[3]*xr[3];
    const float q2 = xr[4]*xr[4] + xr[5]*xr[5] + xr[6]*xr[6];
    const float q3 = xr[7]*xr[7];
    const float qq[4] = {q0, q1, q2, q3};
    float rinv[4];
#pragma unroll
    for (int g = 0; g < 4; ++g) {
      float nm = sqrtf(qq[g]);
      nm = anp[g] * (nm - 1.0f) + 1.0f;
      rinv[g] = 1.0f / (nm + EPSV);
    }
#pragma unroll
    for (int bl = 0; bl < NBL; ++bl) xr[bl] *= rinv[GR[bl]];

    float geo[NBL] = {0, 0, 0, 0, 0, 0, 0, 0};
#pragma unroll
    for (int i = 0; i < 8; ++i) {
#pragma unroll
      for (int k = 0; k < 8; ++k) {
        const int   jj = TT.j[i][k];
        const float sg = TT.s[i][k];
        const int   pp = TT.p[GR[i]][GR[jj]][GR[k]];
        geo[jj] += sg * pw[pp] * y[i][r] * xr[k];
      }
    }

    float mv[NBL];
    float ssum = 0.0f;
#pragma unroll
    for (int bl = 0; bl < NBL; ++bl) {
      const float v = (cw[bl][r] + ((bl == 0) ? gblb : 0.0f) + geo[bl]) * invs2;
      mv[bl] = v;
      ssum += v * v;
    }
    // cross-channel mean of norms (channels live in 8-lane groups)
    float cn = sqrtf(ssum);
    cn += __shfl_xor(cn, 1, 8);
    cn += __shfl_xor(cn, 2, 8);
    cn += __shfl_xor(cn, 4, 8);
    const float scl = lna / (cn * 0.125f + EPSV);

    const long e = e0 + half * 8 + r;
    if (MODE == 0) {
      if (chan_ok && e < (long)n_items) {
        const int rn = row[e];
        float* dst = outp + (long)rn * (FCH * NBL) + nch * NBL;
#pragma unroll
        for (int bl = 0; bl < NBL; ++bl) atomicAdd(dst + bl, scl * mv[bl]);
      }
    } else {
      if (chan_ok && e < (long)n_items) {
        const float* hb = hsrc + e * (FCH * NBL) + nch * NBL;
        float* dst = outp + e * (FCH * NBL) + nch * NBL;
#pragma unroll
        for (int bl = 0; bl < NBL; ++bl) dst[bl] = hb[bl] + scl * mv[bl];
      }
    }
  }
}

// ---------------------------------------------------------------------------
extern "C" void kernel_launch(void* const* d_in, const int* in_sizes, int n_in,
                              void* d_out, int out_size, void* d_ws,
                              size_t ws_size, hipStream_t stream) {
  (void)n_in; (void)out_size; (void)ws_size;
  const float* h  = (const float*)d_in[0];
  const int*   ei = (const int*)d_in[1];
  const int n_edges = in_sizes[1] / 2;
  const int n_nodes = in_sizes[0] / (FCH * NBL);
  const int* row = ei;
  const int* col = ei + n_edges;

  const float* E[10];
  const float* N[10];
  for (int i = 0; i < 10; ++i) {
    E[i] = (const float*)d_in[2 + i];
    N[i] = (const float*)d_in[12 + i];
  }

  float* agg = (float*)d_ws;            // [n_nodes, F, NB] f32 accumulator
  float* out = (float*)d_out;

  {
    const long n = (long)n_nodes * (FCH * NBL);
    int blocks = (int)((n + 255) / 256);
    if (blocks > 4096) blocks = 4096;
    zero_f32_kernel<<<blocks, 256, 0, stream>>>(agg, n);
  }
  {
    const int tiles = (n_edges + 15) / 16;
    const int blocks = (tiles + 7) / 8;
    clifford_block_kernel<0><<<blocks, 256, 0, stream>>>(
        h, row, col, nullptr,
        E[0], E[1], E[2], E[3], E[4], E[5], E[6], E[7], E[8], E[9],
        agg, n_edges);
  }
  {
    const int tiles = (n_nodes + 15) / 16;
    const int blocks = (tiles + 7) / 8;
    clifford_block_kernel<1><<<blocks, 256, 0, stream>>>(
        h, nullptr, nullptr, agg,
        N[0], N[1], N[2], N[3], N[4], N[5], N[6], N[7], N[8], N[9],
        out, n_nodes);
  }
}
